// GCNModel_13804024889623
// MI455X (gfx1250) — matrix operand order repaired and checked by reading, not compiled, not run on hardware
//
#include <hip/hip_runtime.h>

// ---------------------------------------------------------------------------
// CDNA5 (gfx1250) GCN pipeline:
//   deg -> rsqrt -> rownorm -> scatter-add (L2-resident atomics) ->
//   bf16 fragment pre-pack -> WMMA GEMM (4 N-tiles/wave, +bias, ReLU) x2 ->
//   edge dot + sigmoid
// ---------------------------------------------------------------------------

typedef __attribute__((ext_vector_type(16))) __bf16 v16bf;
typedef __attribute__((ext_vector_type(8)))  float  v8f;

// fp32 -> bf16 round-to-nearest-even
__device__ __forceinline__ unsigned short f2bf_bits(float f) {
    union { float f; unsigned u; } in; in.f = f;
    unsigned u = in.u;
    u += 0x7fffu + ((u >> 16) & 1u);
    return (unsigned short)(u >> 16);
}

// ---------------------------------------------------------------------------
__global__ void zero_f32(float* __restrict__ p, int n) {
    int i = blockIdx.x * blockDim.x + threadIdx.x;
    if (i < n) p[i] = 0.0f;
}

__global__ void degree_kernel(const int* __restrict__ src, const int* __restrict__ dst,
                              float* __restrict__ degOut, float* __restrict__ degIn, int E) {
    int e = blockIdx.x * blockDim.x + threadIdx.x;
    if (e < E) {
        atomicAdd(&degOut[src[e]], 1.0f);
        atomicAdd(&degIn[dst[e]], 1.0f);
    }
}

__global__ void rsqrt_inplace(float* __restrict__ d, int n) {
    int i = blockIdx.x * blockDim.x + threadIdx.x;
    if (i < n) d[i] = __frsqrt_rn(fmaxf(d[i], 1.0f));
}

// hn[v,d] = x[v,d] * rinv[v]   (D = 128 -> shift = 7)
__global__ void rownorm_kernel(const float* __restrict__ x, const float* __restrict__ rinv,
                               float* __restrict__ hn, int total, int shift) {
    int i = blockIdx.x * blockDim.x + threadIdx.x;
    if (i < total) hn[i] = x[i] * rinv[i >> shift];
}

// agg[dst[e], d] += hn[src[e], d]   — atomics stay resident in 192MB L2
__global__ void scatter_add_kernel(const float* __restrict__ hn, const int* __restrict__ src,
                                   const int* __restrict__ dst, float* __restrict__ agg,
                                   int E) {
    int i = blockIdx.x * blockDim.x + threadIdx.x;
    int e = i >> 7;
    if (e < E) {
        int d = i & 127;
        atomicAdd(&agg[(dst[e] << 7) + d], hn[(src[e] << 7) + d]);
    }
}

// ---------------------------------------------------------------------------
// Pack A = (agg * rsqrt(deg_in)) into bf16 WMMA A-fragment order (K = 128).
// Fragment layout (ISA 7.12.2, 16-bit A 16x32): per lane l, element e:
//   row = tileM*16 + (l&15)
//   k   = c*32 + (l>=16 ? 8 : 0) + (e<8 ? e : e+8)
// Linear out index: ((tileM*4 + c)*32 + l)*16 + e   -> coalesced stores.
__global__ void pack_a_bf16(const float* __restrict__ A, const float* __restrict__ rinvIn,
                            unsigned short* __restrict__ out, int total) {
    int idx = blockIdx.x * blockDim.x + threadIdx.x;
    if (idx >= total) return;
    int e  = idx & 15;
    int l  = (idx >> 4) & 31;
    int c  = (idx >> 9) & 3;
    int t  = idx >> 11;
    int row = t * 16 + (l & 15);
    int k   = c * 32 + ((l >> 4) ? 8 : 0) + (e < 8 ? e : e + 8);
    out[idx] = f2bf_bits(A[row * 128 + k] * rinvIn[row]);
}

// Pack W [128,N] into bf16 WMMA B-fragment order:
//   col = tileN*16 + (l&15) ;  k = c*32 + (l>=16 ? 8 : 0) + (e<8 ? e : e+8)
__global__ void pack_w_bf16(const float* __restrict__ W, unsigned short* __restrict__ out,
                            int N, int total) {
    int idx = blockIdx.x * blockDim.x + threadIdx.x;
    if (idx >= total) return;
    int e  = idx & 15;
    int l  = (idx >> 4) & 31;
    int c  = (idx >> 9) & 3;
    int tn = idx >> 11;
    int col = tn * 16 + (l & 15);
    int k   = c * 32 + ((l >> 4) ? 8 : 0) + (e < 8 ? e : e + 8);
    out[idx] = f2bf_bits(W[k * N + col]);
}

// ---------------------------------------------------------------------------
// out[M,N] = relu( Apack @ Wpack + bias ),  K = 128 (4 chunks of 32).
// One wave computes a 16x64 strip: 4 N-tiles share each A fragment.
// Requires M % 16 == 0 (M = 50000 = 3125*16).
__global__ void gemm_wmma_packed(const unsigned short* __restrict__ Ap,
                                 const unsigned short* __restrict__ Wp,
                                 const float* __restrict__ bias,
                                 float* __restrict__ out,
                                 int N, int nGroups, int totWaves) {
    const int wave = threadIdx.x >> 5;
    const int lane = threadIdx.x & 31;
    const int t    = blockIdx.x * 4 + wave;
    if (t >= totWaves) return;

    const int tileM  = t / nGroups;
    const int grp    = t - tileM * nGroups;       // group of 4 N-tiles
    const int laneHi = lane >> 4;
    const int lmod   = lane & 15;

    // per-lane fragment base: fragment = 512 ushorts, lane slice = 16 ushorts
    const unsigned short* aBase = Ap + ((size_t)tileM * 4 * 32 + lane) * 16;

    v8f acc[4] = {};
#pragma unroll
    for (int c = 0; c < 4; ++c) {
        const v16bf a = *reinterpret_cast<const v16bf*>(aBase + (size_t)c * 512);
#pragma unroll
        for (int nt = 0; nt < 4; ++nt) {
            const int tileN = grp * 4 + nt;
            const v16bf b = *reinterpret_cast<const v16bf*>(
                Wp + (((size_t)tileN * 4 + c) * 32 + lane) * 16);
            acc[nt] = __builtin_amdgcn_wmma_f32_16x16x32_bf16(
                          false, a, false, b, (short)0, acc[nt], false, false);
        }
    }

#pragma unroll
    for (int nt = 0; nt < 4; ++nt) {
        const int col = (grp * 4 + nt) * 16 + lmod;
        const float bn = bias[col];
#pragma unroll
        for (int r = 0; r < 8; ++r) {
            const int m = tileM * 16 + r + laneHi * 8;   // C/D layout: VGPR r -> M=r (+8)
            const float v = acc[nt][r] + bn;
            out[m * N + col] = v > 0.0f ? v : 0.0f;
        }
    }
}

// score[e] = sigmoid( dot(x[src[e]], x[dst[e]]) ), 64 dims, one wave32 per edge
__global__ void edge_score_kernel(const float* __restrict__ x, const int* __restrict__ src,
                                  const int* __restrict__ dst, float* __restrict__ out, int E) {
    int gid  = blockIdx.x * blockDim.x + threadIdx.x;
    int e    = gid >> 5;
    int lane = threadIdx.x & 31;
    if (e >= E) return;
    const float* xs = x + (size_t)src[e] * 64;
    const float* xt = x + (size_t)dst[e] * 64;
    float acc = xs[lane] * xt[lane] + xs[lane + 32] * xt[lane + 32];
#pragma unroll
    for (int off = 16; off >= 1; off >>= 1)
        acc += __shfl_xor(acc, off, 32);
    if (lane == 0) out[e] = 1.0f / (1.0f + __expf(-acc));
}

// ---------------------------------------------------------------------------
extern "C" void kernel_launch(void* const* d_in, const int* in_sizes, int n_in,
                              void* d_out, int out_size, void* d_ws, size_t ws_size,
                              hipStream_t stream) {
    const float* features = (const float*)d_in[0];   // [50000,128]
    const int*   src      = (const int*)  d_in[1];   // [E]
    const int*   dst      = (const int*)  d_in[2];   // [E]
    /* d_in[3] = edge_type, unused by reference      */
    const float* W1       = (const float*)d_in[4];   // [128,128]
    const float* b1       = (const float*)d_in[5];   // [128]
    const float* W2       = (const float*)d_in[6];   // [128,64]
    const float* b2       = (const float*)d_in[7];   // [64]
    float*       score    = (float*)d_out;           // [E]

    const int Nn = in_sizes[0] / 128;   // 50000 nodes (multiple of 16)
    const int E  = in_sizes[1];         // 800000 edges

    // ---- workspace carve-out (floats), ~77 MB, all regions 32B-aligned ----
    float* ws     = (float*)d_ws;
    float* degOut = ws;                             // Nn   (-> rsqrt(deg_out))
    float* degIn  = degOut + Nn;                    // Nn   (-> rsqrt(deg_in))
    float* hnorm  = degIn  + Nn;                    // Nn*128
    float* agg    = hnorm  + (size_t)Nn * 128;      // Nn*128
    float* h1     = agg    + (size_t)Nn * 128;      // Nn*128
    float* wpackf = h1     + (size_t)Nn * 128;      // 8192 floats (16384 bf16)

    unsigned short* Apack = (unsigned short*)hnorm; // alias: hnorm dead when packed
    unsigned short* Wpack = (unsigned short*)wpackf;
    float* h2 = agg;                                // alias: agg dead during L2 GEMM

    const int B    = 256;
    const int nodF = Nn * 128;
    const int edgF = E * 128;
    const int mT   = Nn / 16;                       // 3125 M-tiles

    // ---- degrees ----
    zero_f32<<<(2 * Nn + B - 1) / B, B, 0, stream>>>(degOut, 2 * Nn);
    degree_kernel<<<(E + B - 1) / B, B, 0, stream>>>(src, dst, degOut, degIn, E);
    rsqrt_inplace<<<(2 * Nn + B - 1) / B, B, 0, stream>>>(degOut, 2 * Nn);

    // ---- layer 1 ----
    rownorm_kernel<<<(nodF + B - 1) / B, B, 0, stream>>>(features, degOut, hnorm, nodF, 7);
    zero_f32<<<(nodF + B - 1) / B, B, 0, stream>>>(agg, nodF);
    scatter_add_kernel<<<(edgF + B - 1) / B, B, 0, stream>>>(hnorm, src, dst, agg, E);
    pack_w_bf16<<<(128 * 128 + B - 1) / B, B, 0, stream>>>(W1, Wpack, 128, 128 * 128);
    pack_a_bf16<<<(nodF + B - 1) / B, B, 0, stream>>>(agg, degIn, Apack, nodF);
    {
        int nGroups = 128 / 64, tot = mT * nGroups;      // 6250 waves
        gemm_wmma_packed<<<(tot + 3) / 4, 128, 0, stream>>>(Apack, Wpack, b1, h1,
                                                            128, nGroups, tot);
    }

    // ---- layer 2 ----
    rownorm_kernel<<<(nodF + B - 1) / B, B, 0, stream>>>(h1, degOut, hnorm, nodF, 7);
    zero_f32<<<(nodF + B - 1) / B, B, 0, stream>>>(agg, nodF);
    scatter_add_kernel<<<(edgF + B - 1) / B, B, 0, stream>>>(hnorm, src, dst, agg, E);
    pack_w_bf16<<<(128 * 64 + B - 1) / B, B, 0, stream>>>(W2, Wpack, 64, 128 * 64);
    pack_a_bf16<<<(nodF + B - 1) / B, B, 0, stream>>>(agg, degIn, Apack, nodF);
    {
        int nGroups = 64 / 64, tot = mT * nGroups;       // 3125 waves
        gemm_wmma_packed<<<(tot + 3) / 4, 128, 0, stream>>>(Apack, Wpack, b2, h2,
                                                            64, nGroups, tot);
    }

    // ---- edge scores ----
    edge_score_kernel<<<((E * 32) + B - 1) / B, B, 0, stream>>>(h2, src, dst, score, E);
}
